// MultiScaleBottleneck_88158498718104
// MI455X (gfx1250) — compile-verified, matrix operand order checked by reading, not run on hardware
//
#include <hip/hip_runtime.h>

#define DEV __device__ __forceinline__

constexpr int Bn   = 16;
constexpr int Tn   = 512;
constexpr int NTOK = Bn * Tn;   // 8192 tokens
constexpr int DM   = 128;
constexpr int DI   = 256;
constexpr int DS   = 16;

typedef __bf16 v16bf __attribute__((ext_vector_type(16)));
typedef float  v8f   __attribute__((ext_vector_type(8)));

DEV float gelu_erf(float x) { return 0.5f * x * (1.f + erff(x * 0.70710678118654752f)); }
DEV float silu_f(float x)   { return x / (1.f + __expf(-x)); }
DEV float softplus_f(float x){ return (x > 20.f) ? x : log1pf(__expf(x)); }

// Build one 16x32 bf16 WMMA fragment half-pair for this lane.
// Branchless: vectorized float4 loads; zero-padding (K==8 only) via selects.
template <int K>
DEV v16bf make_frag(const float* __restrict__ p, int k0, int k1)
{
  int o0 = k0, o1 = k1;
  if constexpr (K % 32 != 0) {            // K==8: keep loads in-bounds
    o0 = (k0 < K) ? k0 : 0;
    o1 = (k1 < K) ? k1 : 0;
  }
  const float4 a0 = *(const float4*)(p + o0);
  const float4 a1 = *(const float4*)(p + o0 + 4);
  const float4 b0 = *(const float4*)(p + o1);
  const float4 b1 = *(const float4*)(p + o1 + 4);
  float va[8] = {a0.x, a0.y, a0.z, a0.w, a1.x, a1.y, a1.z, a1.w};
  float vb[8] = {b0.x, b0.y, b0.z, b0.w, b1.x, b1.y, b1.z, b1.w};
  v16bf f;
#pragma unroll
  for (int j = 0; j < 8; ++j) {
    if constexpr (K % 32 != 0) {
      f[j]     = (__bf16)((k0 + j < K) ? va[j] : 0.f);
      f[j + 8] = (__bf16)((k1 + j < K) ? vb[j] : 0.f);
    } else {
      f[j]     = (__bf16)va[j];
      f[j + 8] = (__bf16)vb[j];
    }
  }
  return f;
}

// =====================================================================
// WMMA GEMM:  C[M=8192, Nout] = A[M,K](row-major, lda) @ W[Nout,K]^T (+bias)
// Optional time-flip (within T=512 segments) on A reads / C writes,
// optional softplus epilogue. One wave owns a 16-row strip, caches the
// A strip as bf16 fragments in VGPRs, sweeps all N tiles.
// =====================================================================
template <int K>
__launch_bounds__(256)
__global__ void gemm_wmma(const float* __restrict__ A, int lda,
                          const float* __restrict__ W,
                          const float* __restrict__ bias,
                          float* __restrict__ C, int ldc,
                          int Nout, int flipA, int flipC, int act)
{
  constexpr int KT = (K + 31) / 32;
  const int lane  = threadIdx.x & 31;
  const int mt    = (blockIdx.x * blockDim.x + threadIdx.x) >> 5;  // 16-row tile
  const int l16   = lane & 15;
  const int lhalf = lane >> 4;
  const int khalf = lhalf * 8;   // 16-bit A/B layout: lane<16 -> K 0..7 & 16..23; lane>=16 -> 8..15 & 24..31

  int arow = mt * 16 + l16;
  if (flipA) arow = (arow & ~(Tn - 1)) + (Tn - 1 - (arow & (Tn - 1)));
  const float* ap = A + (size_t)arow * lda;

  v16bf afrag[KT];
#pragma unroll
  for (int kk = 0; kk < KT; ++kk)
    afrag[kk] = make_frag<K>(ap, kk * 32 + khalf, kk * 32 + 16 + khalf);

  const int ntiles = (Nout + 15) >> 4;
  for (int nt = 0; nt < ntiles; ++nt) {
    const int ncol = nt * 16 + l16;
    // Clamp: garbage in columns >= Nout only affects unstored outputs.
    const int ncc  = (ncol < Nout) ? ncol : (Nout - 1);
    const float* wp = W + (size_t)ncc * K;
    if (ncol + 16 < Nout) __builtin_prefetch(wp + (size_t)16 * K, 0, 1);  // global_prefetch_b8

    v8f acc = {};
#pragma unroll
    for (int kk = 0; kk < KT; ++kk) {
      const v16bf bfrag = make_frag<K>(wp, kk * 32 + khalf, kk * 32 + 16 + khalf);
      acc = __builtin_amdgcn_wmma_f32_16x16x32_bf16(false, afrag[kk], false, bfrag,
                                                    (short)0, acc, false, false);
    }
    if (ncol < Nout) {
      const float bv   = bias ? bias[ncol] : 0.f;
      const int  mbase = mt * 16 + lhalf * 8;   // C/D layout: VGPR r -> M = r (+8 for upper lanes)
#pragma unroll
      for (int r = 0; r < 8; ++r) {
        const int m = mbase + r;
        float v = acc[r] + bv;
        if (act == 1) v = softplus_f(v);
        int cm = m;
        if (flipC) cm = (m & ~(Tn - 1)) + (Tn - 1 - (m & (Tn - 1)));
        C[(size_t)cm * ldc + ncol] = v;
      }
    }
  }
}

// =====================================================================
// LayerNorm over 128 features, wave-per-row. mode 0: dst = LN(src)
// mode 1: dst = res + gelu(LN(src))
// =====================================================================
__launch_bounds__(256)
__global__ void ln_kernel(const float* __restrict__ src,
                          const float* __restrict__ g, const float* __restrict__ bsh,
                          const float* __restrict__ res, float* __restrict__ dst, int mode)
{
  const int lane = threadIdx.x & 31;
  const int row  = blockIdx.x * 8 + (threadIdx.x >> 5);
  const float* sr = src + (size_t)row * DM;
  float v[4];
  float s = 0.f, q = 0.f;
#pragma unroll
  for (int j = 0; j < 4; ++j) {
    v[j] = sr[lane + j * 32];
    s += v[j]; q += v[j] * v[j];
  }
#pragma unroll
  for (int off = 16; off; off >>= 1) { s += __shfl_xor(s, off); q += __shfl_xor(q, off); }
  const float mean = s * (1.f / DM);
  const float var  = q * (1.f / DM) - mean * mean;
  const float rstd = rsqrtf(var + 1e-5f);
#pragma unroll
  for (int j = 0; j < 4; ++j) {
    const int c = lane + j * 32;
    float o = (v[j] - mean) * rstd * g[c] + bsh[c];
    if (mode == 1) o = res[(size_t)row * DM + c] + gelu_erf(o);
    dst[(size_t)row * DM + c] = o;
  }
}

// =====================================================================
// Multi-scale conv branches + BatchNorm(eval) + GELU -> ms (NTOK,128)
// =====================================================================
struct BranchArgs {
  const float* w[4]; const float* bias[4]; const float* gm[4];
  const float* bt[4]; const float* rm[4]; const float* rv[4];
};

__launch_bounds__(256)
__global__ void msconv_kernel(const float* __restrict__ x, BranchArgs ba,
                              float* __restrict__ ms)
{
  const int gid = blockIdx.x * blockDim.x + threadIdx.x;
  if (gid >= NTOK * DM) return;
  const int c = gid & (DM - 1);
  const int row = gid >> 7;
  const int bi = row >> 9;
  const int t  = row & (Tn - 1);
  const int br = c >> 5, oc = c & 31;
  const int k = 2 * br + 1, half = br;

  const float* w = ba.w[br] + (size_t)oc * DM * k;
  float acc = ba.bias[br][oc];
  for (int j = 0; j < k; ++j) {
    const int tt = t + j - half;
    if (tt < 0 || tt >= Tn) continue;
    const float* xr = x + ((size_t)(bi * Tn + tt)) * DM;
    float a = 0.f;
    for (int ic = 0; ic < DM; ++ic) a += xr[ic] * w[ic * k + j];
    acc += a;
  }
  float o = (acc - ba.rm[br][oc]) * rsqrtf(ba.rv[br][oc] + 1e-5f);
  o = o * ba.gm[br][oc] + ba.bt[br][oc];
  ms[(size_t)row * DM + c] = gelu_erf(o);
}

// =====================================================================
// Causal depthwise conv1d (k=4) + SiLU on xi = xz[:, :256]
// =====================================================================
__launch_bounds__(256)
__global__ void dwconv_silu_kernel(const float* __restrict__ xz,
                                   const float* __restrict__ cw,
                                   const float* __restrict__ cb,
                                   float* __restrict__ xi)
{
  const int gid = blockIdx.x * blockDim.x + threadIdx.x;
  if (gid >= NTOK * DI) return;
  const int d = gid & (DI - 1);
  const int row = gid >> 8;
  const int t = row & (Tn - 1);
  const float* w = cw + d * 4;
  float acc = cb[d];
#pragma unroll
  for (int j = 0; j < 4; ++j) {
    const int tt = t - 3 + j;
    if (tt >= 0) acc += xz[(size_t)(row - 3 + j) * (2 * DI) + d] * w[j];
  }
  xi[(size_t)row * DI + d] = silu_f(acc);
}

// =====================================================================
// Selective scan: one block per batch, thread d owns h[d][0..15].
// Per step: B/C (16 each) staged through LDS. Epilogue: +x*D, *silu(z).
// =====================================================================
__launch_bounds__(256)
__global__ void scan_kernel(const float* __restrict__ dt_buf,
                            const float* __restrict__ xi_conv,
                            const float* __restrict__ dbl,
                            const float* __restrict__ xzbuf,
                            const float* __restrict__ A_log,
                            const float* __restrict__ Dp,
                            float* __restrict__ y_gated)
{
  __shared__ float sB[DS], sC[DS];
  const int b = blockIdx.x;
  const int d = threadIdx.x;

  float A[DS], h[DS];
#pragma unroll
  for (int s = 0; s < DS; ++s) { A[s] = -__expf(A_log[d * DS + s]); h[s] = 0.f; }
  const float Dv = Dp[d];

  for (int t = 0; t < Tn; ++t) {
    const size_t row = (size_t)b * Tn + t;
    if (threadIdx.x < DS)            sB[threadIdx.x]      = dbl[row * 40 + 8 + threadIdx.x];
    else if (threadIdx.x < 2 * DS)   sC[threadIdx.x - DS] = dbl[row * 40 + 24 + (threadIdx.x - DS)];
    __syncthreads();

    const float dtv = dt_buf[row * DI + d];
    const float xv  = xi_conv[row * DI + d];
    const float dtx = dtv * xv;
    float y = 0.f;
#pragma unroll
    for (int s = 0; s < DS; ++s) {
      const float dA = __expf(dtv * A[s]);
      h[s] = dA * h[s] + dtx * sB[s];
      y += h[s] * sC[s];
    }
    const float zv = xzbuf[row * (2 * DI) + DI + d];
    y += xv * Dv;
    y_gated[row * DI + d] = y * silu_f(zv);
    __syncthreads();
  }
}

// =====================================================================
extern "C" void kernel_launch(void* const* d_in, const int* in_sizes, int n_in,
                              void* d_out, int out_size, void* d_ws, size_t ws_size,
                              hipStream_t stream)
{
  (void)in_sizes; (void)n_in; (void)out_size; (void)ws_size;
  int i = 0;
  auto in = [&](void) { return (const float*)d_in[i++]; };

  const float* x = in();                       // (B,T,128)
  BranchArgs ba;
  for (int j = 0; j < 4; ++j) {
    ba.w[j] = in(); ba.bias[j] = in(); ba.gm[j] = in();
    ba.bt[j] = in(); ba.rm[j] = in(); ba.rv[j] = in();
  }
  const float* fus_w = in(); const float* fus_b = in();
  const float* fus_g = in(); const float* fus_bt = in();
  const float* norm_g = in(); const float* norm_b = in();

  struct Mp { const float *in_proj,*conv_w,*conv_b,*x_proj,*dt_w,*dt_b,*A_log,*D,*out_proj; };
  struct Lp { Mp m[2]; const float *fw,*fb,*lg,*lb; } L[2];
  for (int l = 0; l < 2; ++l) {
    for (int dir = 0; dir < 2; ++dir) {
      Mp& m = L[l].m[dir];
      m.in_proj = in(); m.conv_w = in(); m.conv_b = in(); m.x_proj = in();
      m.dt_w = in(); m.dt_b = in(); m.A_log = in(); m.D = in(); m.out_proj = in();
    }
    L[l].fw = in(); L[l].fb = in(); L[l].lg = in(); L[l].lb = in();
  }

  float* out = (float*)d_out;

  // workspace carve (floats)
  float* ws = (float*)d_ws;
  float* ms      = ws; ws += (size_t)NTOK * DM;       // 1M
  float* tmpF    = ws; ws += (size_t)NTOK * DM;       // 1M
  float* o_norm  = ws; ws += (size_t)NTOK * DM;       // 1M
  float* xz      = ws; ws += (size_t)NTOK * 2 * DI;   // 4M
  float* xi_conv = ws; ws += (size_t)NTOK * DI;       // 2M
  float* dbl     = ws; ws += (size_t)NTOK * 40;       // 0.33M
  float* dt_buf  = ws; ws += (size_t)NTOK * DI;       // 2M
  float* y_gate  = ws; ws += (size_t)NTOK * DI;       // 2M
  float* comb    = ws; ws += (size_t)NTOK * 2 * DM;   // 2M

  const int GEMM_GRID = NTOK / 16 / 8;   // 512 waves, 8 per block
  const int LN_GRID   = NTOK / 8;

  // 1) multi-scale conv + BN + GELU
  msconv_kernel<<<(NTOK * DM + 255) / 256, 256, 0, stream>>>(x, ba, ms);
  // 2) first fusion: tmpF = ms @ fus_w^T + fus_b ; out = x + gelu(LN(tmpF))
  gemm_wmma<128><<<GEMM_GRID, 256, 0, stream>>>(ms, DM, fus_w, fus_b, tmpF, DM, DM, 0, 0, 0);
  ln_kernel<<<LN_GRID, 256, 0, stream>>>(tmpF, fus_g, fus_bt, x, out, 1);

  for (int l = 0; l < 2; ++l) {
    // pre-norm
    ln_kernel<<<LN_GRID, 256, 0, stream>>>(out, norm_g, norm_b, nullptr, o_norm, 0);

    for (int dir = 0; dir < 2; ++dir) {       // 0 = fwd, 1 = bwd (time-flipped)
      const Mp& m = L[l].m[dir];
      const int flip = dir;
      // in_proj: (N,128)@(128,512)^T -> xz  (bwd reads time-flipped rows)
      gemm_wmma<128><<<GEMM_GRID, 256, 0, stream>>>(o_norm, DM, m.in_proj, nullptr,
                                                    xz, 2 * DI, 2 * DI, flip, 0, 0);
      // causal depthwise conv + silu
      dwconv_silu_kernel<<<(NTOK * DI + 255) / 256, 256, 0, stream>>>(xz, m.conv_w, m.conv_b, xi_conv);
      // x_proj: (N,256)@(256,40)^T -> dbl
      gemm_wmma<256><<<GEMM_GRID, 256, 0, stream>>>(xi_conv, DI, m.x_proj, nullptr,
                                                    dbl, 40, 40, 0, 0, 0);
      // dt_proj + softplus: (N,8)@(8,256)^T + b -> dt_buf
      gemm_wmma<8><<<GEMM_GRID, 256, 0, stream>>>(dbl, 40, m.dt_w, m.dt_b,
                                                  dt_buf, DI, DI, 0, 0, 1);
      // selective scan + D skip + z gate
      scan_kernel<<<Bn, DI, 0, stream>>>(dt_buf, xi_conv, dbl, xz, m.A_log, m.D, y_gate);
      // out_proj: (N,256)@(256,128)^T -> comb columns [dir*128, dir*128+128)
      gemm_wmma<256><<<GEMM_GRID, 256, 0, stream>>>(y_gate, DI, m.out_proj, nullptr,
                                                    comb + dir * DM, 2 * DM, DM, 0, flip, 0);
    }
    // layer fusion: (N,256)@(256,128)^T + b -> tmpF ; out = out + gelu(LN(tmpF))
    gemm_wmma<256><<<GEMM_GRID, 256, 0, stream>>>(comb, 2 * DM, L[l].fw, L[l].fb,
                                                  tmpF, DM, DM, 0, 0, 0);
    ln_kernel<<<LN_GRID, 256, 0, stream>>>(tmpF, L[l].lg, L[l].lb, out, out, 1);
  }
}
